// RelativeMultiHeadAttention_47433618817567
// MI455X (gfx1250) — matrix operand order repaired and checked by reading, not compiled, not run on hardware
//
#include <hip/hip_runtime.h>
#include <hip/hip_bf16.h>
#include <math.h>

typedef __attribute__((ext_vector_type(16))) _Float16 v16h;
typedef __attribute__((ext_vector_type(8)))  _Float16 v8h;
typedef __attribute__((ext_vector_type(8)))  float    v8f;

#define Bc 8
#define Tc 1024
#define Dc 512
#define Hc 8
#define DHc 64

// ---------- WMMA fragment helpers (CDNA5 16x16x32 f16 layouts) ----------
// A (16x32, MxK): lane L<16 holds row M=L, K = {k0..k0+7, k0+16..k0+23};
//                 lane L>=16 holds row M=L-16, K = {k0+8..15, k0+24..31}.
__device__ inline v16h ldA(const _Float16* rowptr, int half) {
  union { v16h v; v8h h[2]; } u;
  u.h[0] = *(const v8h*)(rowptr + half * 8);
  u.h[1] = *(const v8h*)(rowptr + 16 + half * 8);
  return u.v;
}
// B (32x16, KxN), source stored N-major (src[n*ld + k]):
// lane L<16 holds col N=L, K = {k0..k0+7, k0+16..k0+23}; lane L>=16 holds col N=L-16, K = {k0+8..15, k0+24..31}.
__device__ inline v16h ldB(const _Float16* colptr, int half) {
  union { v16h v; v8h h[2]; } u;
  u.h[0] = *(const v8h*)(colptr + half * 8);
  u.h[1] = *(const v8h*)(colptr + 16 + half * 8);
  return u.v;
}
__device__ inline v8f wmma16(v16h a, v16h b, v8f c) {
  return __builtin_amdgcn_wmma_f32_16x16x32_f16(false, a, false, b, (short)0, c,
                                                false, false);
}

// ---------- fp32 -> f16 convert ----------
__global__ void cvt_kernel(const float* __restrict__ in, _Float16* __restrict__ out, int n) {
  int i = blockIdx.x * 256 + threadIdx.x;
  if (i < n) out[i] = (_Float16)in[i];
}

// ---------- LayerNorm (one wave32 per 512-wide row) ----------
__global__ __launch_bounds__(32) void ln_kernel(const float* __restrict__ Q,
                                                const float* __restrict__ gamma,
                                                const float* __restrict__ beta,
                                                _Float16* __restrict__ out) {
  int row = blockIdx.x;
  int lane = threadIdx.x;
  const float* src = Q + (size_t)row * Dc;
  float v[16];
  float s = 0.f;
  for (int i = 0; i < 16; ++i) { v[i] = src[lane + i * 32]; s += v[i]; }
  for (int o = 16; o; o >>= 1) s += __shfl_xor(s, o, 32);
  float mu = s * (1.f / Dc);
  float q = 0.f;
  for (int i = 0; i < 16; ++i) { float d = v[i] - mu; q += d * d; }
  for (int o = 16; o; o >>= 1) q += __shfl_xor(q, o, 32);
  float rs = rsqrtf(q * (1.f / Dc) + 1e-5f);
  for (int i = 0; i < 16; ++i) {
    int col = lane + i * 32;
    out[(size_t)row * Dc + col] = (_Float16)((v[i] - mu) * rs * gamma[col] + beta[col]);
  }
}

// ---------- tile store epilogues ----------
// mode 0: f16 -> [B,H,T,DH] ; mode 1: f16 transposed -> [B,H,DH,T]
// mode 2: f32 = acc + bias[c] + resid[r,c]  (output projection + residual)
__device__ inline void store_tile(v8f c, int r0, int n0, int lr, int half,
                                  _Float16* __restrict__ out16,
                                  float* __restrict__ out32,
                                  const float* __restrict__ bias,
                                  const float* __restrict__ resid, int mode) {
  if (mode == 2) {
    int col = n0 + lr;
    float bz = bias[col];
    for (int i = 0; i < 8; ++i) {
      int r = r0 + half * 8 + i;
      size_t off = (size_t)r * Dc + col;
      out32[off] = c[i] + bz + resid[off];
    }
  } else {
    int bidx = r0 >> 10, t0 = r0 & 1023;
    int h = n0 >> 6, dh0 = n0 & 63;
    if (mode == 0) {
      for (int i = 0; i < 8; ++i) {
        int m = i + half * 8;
        out16[((size_t)((bidx * Hc + h) * Tc + t0 + m)) * DHc + dh0 + lr] = (_Float16)c[i];
      }
    } else {  // mode 1: v transposed [B,H,DH,T]
      v8h pk;
      for (int i = 0; i < 8; ++i) pk[i] = (_Float16)c[i];
      *(v8h*)(out16 + ((size_t)((bidx * Hc + h) * DHc + dh0 + lr)) * Tc + t0 + half * 8) = pk;
    }
  }
}

// ---------- GEMM: C[M,N] = A[M,K] * W[N,K]^T ----------
// 32x32 per wave (4 C tiles): 2 A-frags + 2 B-frags feed 4 WMMAs per K-step,
// doubling FLOPs per byte of VGPR fetch vs a 16x16 tile. Block = 4 waves = 64x64.
__global__ __launch_bounds__(128) void gemm_kernel(const _Float16* __restrict__ A,
                                                   const _Float16* __restrict__ W,
                                                   _Float16* __restrict__ out16,
                                                   float* __restrict__ out32,
                                                   const float* __restrict__ bias,
                                                   const float* __restrict__ resid,
                                                   int K, int mode) {
  int wave = threadIdx.x >> 5;
  int lane = threadIdx.x & 31;
  int lr = lane & 15, half = lane >> 4;
  int r0 = blockIdx.x * 64 + (wave >> 1) * 32;
  int n0 = blockIdx.y * 64 + (wave & 1) * 32;
  const _Float16* arow0 = A + (size_t)(r0 + lr) * K;
  const _Float16* arow1 = A + (size_t)(r0 + 16 + lr) * K;
  const _Float16* wcol0 = W + (size_t)(n0 + lr) * K;
  const _Float16* wcol1 = W + (size_t)(n0 + 16 + lr) * K;
  v8f c00 = {0.f, 0.f, 0.f, 0.f, 0.f, 0.f, 0.f, 0.f};
  v8f c01 = c00, c10 = c00, c11 = c00;
  for (int k0 = 0; k0 < K; k0 += 32) {
    v16h a0 = ldA(arow0 + k0, half);
    v16h a1 = ldA(arow1 + k0, half);
    v16h b0 = ldB(wcol0 + k0, half);
    v16h b1 = ldB(wcol1 + k0, half);
    c00 = wmma16(a0, b0, c00);
    c01 = wmma16(a0, b1, c01);
    c10 = wmma16(a1, b0, c10);
    c11 = wmma16(a1, b1, c11);
  }
  store_tile(c00, r0,      n0,      lr, half, out16, out32, bias, resid, mode);
  store_tile(c01, r0,      n0 + 16, lr, half, out16, out32, bias, resid, mode);
  store_tile(c10, r0 + 16, n0,      lr, half, out16, out32, bias, resid, mode);
  store_tile(c11, r0 + 16, n0 + 16, lr, half, out16, out32, bias, resid, mode);
}

// ---------- Flash attention with relative positions ----------
// grid = B*H*(T/128) blocks of 256 threads (8 waves). Wave w owns query rows
// [tw, tw+16). All score tiles computed TRANSPOSED (M=s, N=t) so each lane's
// column is one query row: softmax stats = in-lane reduce + one xor-16 shuffle.
__global__ __launch_bounds__(256) void attn_kernel(const _Float16* __restrict__ qg,
                                                   const _Float16* __restrict__ kg,
                                                   const _Float16* __restrict__ vTg,
                                                   const _Float16* __restrict__ tab,
                                                   float* __restrict__ ctx32,
                                                   _Float16* __restrict__ ctx16) {
  __shared__ __align__(32) float    smemE[8][48 * 16];  // banded rel scores E^T[j][t]
  __shared__ __align__(32) _Float16 smemP[8][16 * 32];  // P^T staged as B-fragments
  int wave = threadIdx.x >> 5;
  int lane = threadIdx.x & 31;
  int lr = lane & 15, half = lane >> 4;
  int bh = blockIdx.x >> 3;
  int tblk = blockIdx.x & 7;
  int tw = tblk * 128 + wave * 16;
  const _Float16* qh = qg + (size_t)bh * Tc * DHc;
  const _Float16* kh = kg + (size_t)bh * Tc * DHc;
  const _Float16* vh = vTg + (size_t)bh * DHc * Tc;
  float* El = smemE[wave];
  _Float16* Pl = smemP[wave];

  // q B-fragments (K = head dim, N = 16 query rows) — reused every iteration
  const _Float16* qcol = qh + (size_t)(tw + lr) * DHc;
  v16h bq0 = ldB(qcol, half);
  v16h bq1 = ldB(qcol + 32, half);

  v8f acc[4];
  for (int d = 0; d < 4; ++d)
    for (int i = 0; i < 8; ++i) acc[d][i] = 0.f;
  float mrow = -1e30f, lrow = 0.f;

  for (int s0 = 0; s0 <= tw + 15; s0 += 32) {
    // speculative prefetch of next chunk (global_prefetch_b8; never faults)
    __builtin_prefetch(kh + (size_t)(s0 + 32 + lr) * DHc, 0, 1);
    __builtin_prefetch(vh + (size_t)lr * Tc + s0 + 32, 0, 1);

    // ---- content scores S^T : 2 M-tiles (ds 0..15, 16..31), K=64 ----
    v8f c0 = {0.f, 0.f, 0.f, 0.f, 0.f, 0.f, 0.f, 0.f};
    v8f c1 = c0;
    const _Float16* kr0 = kh + (size_t)(s0 + lr) * DHc;
    const _Float16* kr1 = kh + (size_t)(s0 + 16 + lr) * DHc;
    c0 = wmma16(ldA(kr0, half), bq0, c0);
    c0 = wmma16(ldA(kr0 + 32, half), bq1, c0);
    c1 = wmma16(ldA(kr1, half), bq0, c1);
    c1 = wmma16(ldA(kr1 + 32, half), bq1, c1);

    // ---- relative scores: banded GEMM over 48 lookup rows ----
    // rel[t,s] = q[t].table[s-t+1023]; j = ds-dt+15 in [0,46], base in [0,1023]
    int base = s0 - tw + 1008;
    for (int mt = 0; mt < 3; ++mt) {
      const _Float16* trow = tab + (size_t)(base + mt * 16 + lr) * DHc;
      v8f e = {0.f, 0.f, 0.f, 0.f, 0.f, 0.f, 0.f, 0.f};
      e = wmma16(ldA(trow, half), bq0, e);
      e = wmma16(ldA(trow + 32, half), bq1, e);
      for (int i = 0; i < 8; ++i)
        El[(mt * 16 + half * 8 + i) * 16 + lr] = e[i];
    }

    // ---- combine + causal mask + online softmax ----
    float sv[16];
    for (int i = 0; i < 8; ++i) {
      int ds = half * 8 + i;
      float s = (c0[i] + El[(ds - lr + 15) * 16 + lr]) * 0.125f;
      sv[i] = (s0 + ds > tw + lr) ? -1e30f : s;
    }
    for (int i = 0; i < 8; ++i) {
      int ds = 16 + half * 8 + i;
      float s = (c1[i] + El[(ds - lr + 15) * 16 + lr]) * 0.125f;
      sv[8 + i] = (s0 + ds > tw + lr) ? -1e30f : s;
    }
    float cm = sv[0];
    for (int i = 1; i < 16; ++i) cm = fmaxf(cm, sv[i]);
    cm = fmaxf(cm, __shfl_xor(cm, 16, 32));
    float mnew = fmaxf(mrow, cm);
    float alpha = expf(mrow - mnew);
    float psum = 0.f;
    v8h p0, p1;
    for (int i = 0; i < 8; ++i) { float p = expf(sv[i] - mnew);     psum += p; p0[i] = (_Float16)p; }
    for (int i = 0; i < 8; ++i) { float p = expf(sv[8 + i] - mnew); psum += p; p1[i] = (_Float16)p; }
    psum += __shfl_xor(psum, 16, 32);
    lrow = lrow * alpha + psum;
    mrow = mnew;
    for (int d = 0; d < 4; ++d)
      for (int i = 0; i < 8; ++i) acc[d][i] *= alpha;

    // stage P^T (Pl[t][ds]) so it reloads as a native B fragment (K = s)
    *(v8h*)(Pl + lr * 32 + half * 8) = p0;
    *(v8h*)(Pl + lr * 32 + 16 + half * 8) = p1;
    v16h bp = ldB(Pl + lr * 32, half);

    // ---- ctx^T += v^T(dx16, K=32 keys) * P^T ----
    for (int d = 0; d < 4; ++d) {
      const _Float16* vrow = vh + (size_t)(d * 16 + lr) * Tc + s0;
      acc[d] = wmma16(ldA(vrow, half), bp, acc[d]);
    }
  }

  // ---- epilogue: normalize, store ctx fp32 (residual) + f16 (out-proj A) ----
  float inv = 1.f / lrow;
  int b = bh >> 3, h = bh & 7;
  int t = tw + lr;
  size_t rbase = ((size_t)(b * Tc + t)) * Dc + h * DHc + half * 8;
  for (int d = 0; d < 4; ++d) {
    v8f o;
    v8h o16;
    for (int i = 0; i < 8; ++i) {
      float x = acc[d][i] * inv;
      o[i] = x;
      o16[i] = (_Float16)x;
    }
    *(v8f*)(ctx32 + rbase + d * 16) = o;
    *(v8h*)(ctx16 + rbase + d * 16) = o16;
  }
}

extern "C" void kernel_launch(void* const* d_in, const int* in_sizes, int n_in,
                              void* d_out, int out_size, void* d_ws, size_t ws_size,
                              hipStream_t stream) {
  const float* Q     = (const float*)d_in[0];
  const float* Kin   = (const float*)d_in[1];
  const float* Vin   = (const float*)d_in[2];
  const float* table = (const float*)d_in[3];
  const float* Wq    = (const float*)d_in[4];
  const float* Wk    = (const float*)d_in[5];
  const float* Wv    = (const float*)d_in[6];
  const float* Wo    = (const float*)d_in[7];
  const float* bo    = (const float*)d_in[8];
  const float* gamma = (const float*)d_in[9];
  const float* beta  = (const float*)d_in[10];
  (void)in_sizes; (void)n_in; (void)out_size; (void)ws_size;

  char* ws = (char*)d_ws;
  size_t off = 0;
  auto alloc = [&](size_t bytes) {
    size_t cur = off;
    off += (bytes + 255) & ~(size_t)255;
    return cur;
  };
  const size_t BT = (size_t)Bc * Tc;           // 8192
  _Float16* Qn16  = (_Float16*)(ws + alloc(BT * Dc * 2));
  _Float16* K16   = (_Float16*)(ws + alloc(BT * Dc * 2));
  _Float16* V16   = (_Float16*)(ws + alloc(BT * Dc * 2));
  _Float16* Wq16  = (_Float16*)(ws + alloc((size_t)Dc * Dc * 2));
  _Float16* Wk16  = (_Float16*)(ws + alloc((size_t)Dc * Dc * 2));
  _Float16* Wv16  = (_Float16*)(ws + alloc((size_t)Dc * Dc * 2));
  _Float16* Wo16  = (_Float16*)(ws + alloc((size_t)Dc * Dc * 2));
  _Float16* tab16 = (_Float16*)(ws + alloc((size_t)(2 * Tc - 1) * DHc * 2));
  _Float16* q16   = (_Float16*)(ws + alloc(BT * Dc * 2));  // [B,H,T,DH]
  _Float16* k16   = (_Float16*)(ws + alloc(BT * Dc * 2));  // [B,H,T,DH]
  _Float16* vT16  = (_Float16*)(ws + alloc(BT * Dc * 2));  // [B,H,DH,T]
  float*    ctx32 = (float*)   (ws + alloc(BT * Dc * 4));
  _Float16* ctx16 = (_Float16*)(ws + alloc(BT * Dc * 2));

  auto cvt = [&](const float* src, _Float16* dst, int n) {
    cvt_kernel<<<(n + 255) / 256, 256, 0, stream>>>(src, dst, n);
  };
  cvt(Kin, K16, (int)(BT * Dc));
  cvt(Vin, V16, (int)(BT * Dc));
  cvt(Wq, Wq16, Dc * Dc);
  cvt(Wk, Wk16, Dc * Dc);
  cvt(Wv, Wv16, Dc * Dc);
  cvt(Wo, Wo16, Dc * Dc);
  cvt(table, tab16, (2 * Tc - 1) * DHc);

  ln_kernel<<<(int)BT, 32, 0, stream>>>(Q, gamma, beta, Qn16);

  dim3 ggrid((int)(BT / 64), Dc / 64);
  gemm_kernel<<<ggrid, 128, 0, stream>>>(Qn16, Wq16, q16, nullptr, nullptr, nullptr, Dc, 0);
  gemm_kernel<<<ggrid, 128, 0, stream>>>(K16, Wk16, k16, nullptr, nullptr, nullptr, Dc, 0);
  gemm_kernel<<<ggrid, 128, 0, stream>>>(V16, Wv16, vT16, nullptr, nullptr, nullptr, Dc, 1);

  attn_kernel<<<Bc * Hc * (Tc / 128), 256, 0, stream>>>(q16, k16, vT16, tab16, ctx32, ctx16);

  gemm_kernel<<<ggrid, 128, 0, stream>>>(ctx16, Wo16, nullptr, (float*)d_out, bo, ctx32, Dc, 2);
}
